// GATKANLayer_32392643346863
// MI455X (gfx1250) — compile-verified, matrix operand order checked by reading, not run on hardware
//
#include <hip/hip_runtime.h>

// ---------------- WMMA fragment helpers (gfx1250, wave32, 16x16x32 f16) ----
typedef __attribute__((ext_vector_type(16))) _Float16 v16h;
typedef __attribute__((ext_vector_type(8)))  float    v8f;

// K index of fragment element e (0..15) for A (16x32) / B (32x16) 16-bit frags.
// Per ISA 7.12.2: VGPR0..3 hold K 0..7 (lanes 0-15) / 8..15 (lanes 16-31),
// VGPR4..7 hold K 16..23 / 24..31. Element e = 2*vgpr + slot.
__device__ __forceinline__ int frag_k(int e, int half) {
  int v = e >> 1, s = e & 1;
  return ((v & 4) << 2) + (half << 3) + ((v & 3) << 1) + s;
}

__device__ __forceinline__ v16h afrag_f16(const _Float16* A, int lda, int k0) {
  int lane = threadIdx.x & 31, m = lane & 15, half = lane >> 4;
  v16h a;
#pragma unroll
  for (int e = 0; e < 16; ++e) a[e] = A[m * lda + k0 + frag_k(e, half)];
  return a;
}

// Pre-swizzled B fragments: ((nt*ktiles + kt)*32 + lane)*16 + e, f16, 32B/lane
__device__ __forceinline__ v16h bfrag_pre(const _Float16* P, int ktiles, int kt,
                                          int nt) {
  int lane = threadIdx.x & 31;
  return *(const v16h*)(P + (((long)nt * ktiles + kt) * 32 + lane) * 16);
}

__device__ __forceinline__ v8f cfrag_ld(const float* Cp, int ldc, int n0) {
  int lane = threadIdx.x & 31, n = lane & 15, half = lane >> 4;
  v8f c;
#pragma unroll
  for (int r = 0; r < 8; ++r) c[r] = Cp[(r + half * 8) * ldc + n0 + n];
  return c;
}

__device__ __forceinline__ void dfrag_st(float* Y, int ldy, int n0, v8f d) {
  int lane = threadIdx.x & 31, n = lane & 15, half = lane >> 4;
#pragma unroll
  for (int r = 0; r < 8; ++r) Y[(r + half * 8) * ldy + n0 + n] = d[r];
}

__device__ __forceinline__ void dfrag_st_h(_Float16* Y, int ldy, int n0, v8f d) {
  int lane = threadIdx.x & 31, n = lane & 15, half = lane >> 4;
#pragma unroll
  for (int r = 0; r < 8; ++r)
    Y[(r + half * 8) * ldy + n0 + n] = (_Float16)d[r];
}

__device__ __forceinline__ v8f wmma16(v16h a, v16h b, v8f c) {
  return __builtin_amdgcn_wmma_f32_16x16x32_f16(false, a, false, b, (short)0, c,
                                                false, false);
}

__device__ __forceinline__ float leaky(float x) { return x > 0.f ? x : 0.01f * x; }

// order-preserving float<->uint map for atomic max
__device__ __forceinline__ unsigned fmapu(float f) {
  unsigned b = __float_as_uint(f);
  return (b & 0x80000000u) ? ~b : (b | 0x80000000u);
}
__device__ __forceinline__ float funmap(unsigned u) {
  unsigned b = (u & 0x80000000u) ? (u & 0x7FFFFFFFu) : ~u;
  return __uint_as_float(b);
}

// ------------------------------------------------------------------ kernels
__global__ void k_zero(float* p, long n) {
  long i = (long)blockIdx.x * blockDim.x + threadIdx.x;
  if (i < n) p[i] = 0.f;
}

// Pre-swizzle one weight matrix into f16 B-fragment layout.
// cm==0: src is B[k][n], ld = row stride.  cm==1: src is C[n][k], ld = K.
__global__ void k_prep(const float* __restrict__ src, _Float16* __restrict__ dst,
                       int ktiles, int ntiles, int ld, int cm) {
  long idx = (long)blockIdx.x * blockDim.x + threadIdx.x;
  long total = (long)ktiles * ntiles * 512;
  if (idx >= total) return;
  int e = (int)(idx & 15);
  int lane = (int)((idx >> 4) & 31);
  long tile = idx >> 9;
  int kt = (int)(tile % ktiles), nt = (int)(tile / ktiles);
  int k = kt * 32 + frag_k(e, lane >> 4);
  int n = nt * 16 + (lane & 15);
  float v = cm ? src[(long)n * ld + k] : src[(long)k * ld + n];
  dst[idx] = (_Float16)v;
}

// f_ni = nf @ W_ni, f_nj = nf @ W_nj   [N,128]x[128,128] (A frags reused)
__global__ __launch_bounds__(256) void k_node_lin(
    const float* __restrict__ nf, const _Float16* __restrict__ pWni,
    const _Float16* __restrict__ pWnj, float* __restrict__ fni,
    float* __restrict__ fnj, int N) {
  __shared__ _Float16 X[16 * 128];
  int n0 = blockIdx.x * 16, tid = threadIdx.x;
  for (int idx = tid; idx < 16 * 128; idx += 256) {
    int r = idx >> 7, c = idx & 127;
    int n = n0 + r; if (n >= N) n = N - 1;
    X[idx] = (_Float16)nf[(long)n * 128 + c];
  }
  __syncthreads();
  int w = tid >> 5;
  int lane = tid & 31, nn = lane & 15, half = lane >> 4;
  v8f a0 = {}, a1 = {};
#pragma unroll
  for (int k0 = 0; k0 < 128; k0 += 32) {
    v16h a = afrag_f16(X, 128, k0);
    a0 = wmma16(a, bfrag_pre(pWni, 4, k0 >> 5, w), a0);
    a1 = wmma16(a, bfrag_pre(pWnj, 4, k0 >> 5, w), a1);
  }
#pragma unroll
  for (int r = 0; r < 8; ++r) {
    int row = n0 + r + half * 8;
    if (row < N) {
      fni[(long)row * 128 + w * 16 + nn] = a0[r];
      fnj[(long)row * 128 + w * 16 + nn] = a1[r];
    }
  }
}

// mean-aggregation accumulators: agg[dst] += edge_feats (float4), deg[dst] += 1
__global__ void k_agg(const float* __restrict__ ef, const int* __restrict__ dst,
                      float* __restrict__ agg, float* __restrict__ deg, int E) {
  long idx = (long)blockIdx.x * blockDim.x + threadIdx.x;
  if (idx >= (long)E * 16) return;
  int e = (int)(idx >> 4), q = (int)(idx & 15);
  int d = dst[e];
  float4 v = ((const float4*)ef)[(long)e * 16 + q];
  float* a = &agg[(long)d * 64 + q * 4];
  atomicAdd(a + 0, v.x); atomicAdd(a + 1, v.y);
  atomicAdd(a + 2, v.z); atomicAdd(a + 3, v.w);
  if (q == 0) atomicAdd(&deg[d], 1.0f);
}

// fused edge path: gather + GEMM + KAN + leaky + scores/segmax + f_sum + out_edge
__global__ __launch_bounds__(256) void k_edge(
    const float* __restrict__ ef, const int* __restrict__ src,
    const int* __restrict__ dst, const float* __restrict__ fni,
    const float* __restrict__ fnj, const _Float16* __restrict__ pWfij,
    const _Float16* __restrict__ pEk0, const _Float16* __restrict__ pEk1,
    const float* __restrict__ ekb, const float* __restrict__ be,
    const float* __restrict__ attn,  // [4][32]
    const _Float16* __restrict__ pOe0, const _Float16* __restrict__ pOe1,
    const float* __restrict__ oeb, float* __restrict__ scores,
    unsigned* __restrict__ smaxU, float* __restrict__ out_edge, int E) {
  __shared__ float Xf[16 * 128];          // f accumulator / activations (C/D)
  __shared__ _Float16 Efh[16 * 64];       // edge feats (A, f16)
  __shared__ _Float16 CSh[2 * 16 * 128];  // cos|sin f16, later A[16][256]
  __shared__ float FS[16 * 32];
  __shared__ int sidx[16], didx[16];
  int e0 = blockIdx.x * 16, tid = threadIdx.x;
  if (tid < 16) {
    int e = e0 + tid; if (e >= E) e = E - 1;
    sidx[tid] = src[e]; didx[tid] = dst[e];
  }
  __syncthreads();
  for (int idx = tid; idx < 16 * 128; idx += 256) {
    int r = idx >> 7, c = idx & 127;
    Xf[idx] = fni[(long)sidx[r] * 128 + c] + fnj[(long)didx[r] * 128 + c];
  }
  for (int idx = tid; idx < 16 * 64; idx += 256) {
    int r = idx >> 6, c = idx & 63;
    int e = e0 + r; if (e >= E) e = E - 1;
    Efh[idx] = (_Float16)ef[(long)e * 64 + c];
  }
  __syncthreads();
  int w = tid >> 5, c0 = w * 16;
  int lane = tid & 31, nn = lane & 15, half = lane >> 4;
  {  // f += edge_feats @ W_fij  (C initialized with gathered sums)
    v8f c = cfrag_ld(Xf, 128, c0);
#pragma unroll
    for (int k0 = 0; k0 < 64; k0 += 32)
      c = wmma16(afrag_f16(Efh, 64, k0), bfrag_pre(pWfij, 2, k0 >> 5, w), c);
    dfrag_st(Xf, 128, c0, c);
  }
  __syncthreads();
  __builtin_prefetch(pEk0, 0, 3);
  for (int idx = tid; idx < 16 * 128; idx += 256) {
    float s, cv; __sincosf(Xf[idx], &s, &cv);
    CSh[idx] = (_Float16)cv; CSh[2048 + idx] = (_Float16)s;
  }
  __syncthreads();
  {  // edge KAN: y = cos@C0^T + sin@C1^T + ek_b + bias_edge; leaky
    float b0 = ekb[c0 + nn] + be[c0 + nn];
    v8f c;
#pragma unroll
    for (int r = 0; r < 8; ++r) c[r] = b0;
#pragma unroll
    for (int k0 = 0; k0 < 128; k0 += 32) {
      c = wmma16(afrag_f16(CSh, 128, k0), bfrag_pre(pEk0, 4, k0 >> 5, w), c);
      c = wmma16(afrag_f16(CSh + 2048, 128, k0), bfrag_pre(pEk1, 4, k0 >> 5, w),
                 c);
    }
#pragma unroll
    for (int r = 0; r < 8; ++r) c[r] = leaky(c[r]);
    dfrag_st(Xf, 128, c0, c);
  }
  __syncthreads();
  if (tid < 64) {  // waves 0-1 (uniform): attention scores + segment max
    int r = tid >> 2, h = tid & 3;
    float s = 0.f;
    for (int j = 0; j < 32; ++j) s += Xf[r * 128 + h * 32 + j] * attn[h * 32 + j];
    int e = e0 + r;
    if (e < E) {
      scores[(long)e * 4 + h] = s;
      atomicMax(&smaxU[(long)didx[r] * 4 + h], fmapu(s));
    }
  }
  for (int idx = tid; idx < 16 * 32; idx += 256) {
    int r = idx >> 5, j = idx & 31;
    FS[idx] = Xf[r * 128 + j] + Xf[r * 128 + 32 + j] + Xf[r * 128 + 64 + j] +
              Xf[r * 128 + 96 + j];
  }
  __syncthreads();
  // out-edge KAN as GEMM with K=256 (cos(i,g) || sin(i,g)); A f16 into LDS
  for (int idx = tid; idx < 16 * 256; idx += 256) {
    int r = idx >> 8, k = idx & 255;
    int kk = k & 127, i = kk >> 2, g = kk & 3;
    float s, cv; __sincosf(FS[r * 32 + i] * (float)(g + 1), &s, &cv);
    CSh[idx] = (_Float16)((k < 128) ? cv : s);
  }
  __syncthreads();
  if (w < 2) {  // 2 output tiles (O=32); wave-uniform branch
    int o0 = w * 16;
    float b0 = oeb[o0 + nn];
    v8f c;
#pragma unroll
    for (int r = 0; r < 8; ++r) c[r] = b0;
#pragma unroll
    for (int k0 = 0; k0 < 256; k0 += 32) {
      const _Float16* B = (k0 < 128) ? pOe0 : pOe1;
      c = wmma16(afrag_f16(CSh, 256, k0), bfrag_pre(B, 4, (k0 & 127) >> 5, w), c);
    }
#pragma unroll
    for (int r = 0; r < 8; ++r) {
      int e = e0 + r + half * 8;
      if (e < E) out_edge[(long)e * 32 + o0 + nn] = c[r];
    }
  }
}

// softmax pass 2: e = exp(s - max), segment sum
__global__ void k_softmax2(const int* __restrict__ dst,
                           const unsigned* __restrict__ smaxU,
                           float* __restrict__ scores, float* __restrict__ ssum,
                           int E) {
  long idx = (long)blockIdx.x * blockDim.x + threadIdx.x;
  if (idx >= (long)E * 4) return;
  int e = (int)(idx >> 2), h = (int)(idx & 3);
  int d = dst[e];
  float ex = __expf(scores[idx] - funmap(smaxU[(long)d * 4 + h]));
  scores[idx] = ex;
  atomicAdd(&ssum[(long)d * 4 + h], ex);
}

// fused node path: merged -> node KAN (O=192) -> h GEMM (O=512)
__global__ __launch_bounds__(256) void k_node_kan(
    const float* __restrict__ nf, const float* __restrict__ agg,
    const float* __restrict__ deg, const _Float16* __restrict__ pNk0,
    const _Float16* __restrict__ pNk1, const float* __restrict__ nkb,
    const _Float16* __restrict__ pWnode, const float* __restrict__ bnode,
    float* __restrict__ h_ws, int N) {
  __shared__ float M[16 * 192];           // merged (f32, consumed by sincos)
  __shared__ _Float16 CSh[2 * 16 * 192];  // cos|sin (A, f16)
  __shared__ _Float16 Mh[16 * 192];       // KAN output (A for h GEMM, f16)
  int n0 = blockIdx.x * 16, tid = threadIdx.x;
  for (int idx = tid; idx < 16 * 192; idx += 256) {
    int r = idx / 192, c = idx % 192;
    int n = n0 + r; if (n >= N) n = N - 1;
    float v;
    if (c < 128) v = nf[(long)n * 128 + c];
    else v = agg[(long)n * 64 + (c - 128)] / fmaxf(deg[n], 1.0f);
    M[idx] = v;
  }
  __syncthreads();
  for (int idx = tid; idx < 16 * 192; idx += 256) {
    float s, cv; __sincosf(M[idx], &s, &cv);
    CSh[idx] = (_Float16)cv; CSh[16 * 192 + idx] = (_Float16)s;
  }
  __syncthreads();
  int w = tid >> 5, lane = tid & 31, nn = lane & 15, half = lane >> 4;
  for (int t = 0; t < 2; ++t) {  // 12 output tiles over 8 waves
    int tile = w + t * 8;
    if (tile < 12) {
      int o0 = tile * 16;
      float b0 = nkb[o0 + nn];
      v8f c;
#pragma unroll
      for (int r = 0; r < 8; ++r) c[r] = b0;
#pragma unroll
      for (int k0 = 0; k0 < 192; k0 += 32) {
        c = wmma16(afrag_f16(CSh, 192, k0), bfrag_pre(pNk0, 6, k0 >> 5, tile), c);
        c = wmma16(afrag_f16(CSh + 16 * 192, 192, k0),
                   bfrag_pre(pNk1, 6, k0 >> 5, tile), c);
      }
      dfrag_st_h(Mh, 192, o0, c);
    }
  }
  __syncthreads();
  for (int t = 0; t < 4; ++t) {  // h = Mh @ W_node + b_node, 32 tiles
    int nt = w + t * 8, o0 = nt * 16;
    float b0 = bnode[o0 + nn];
    v8f c;
#pragma unroll
    for (int r = 0; r < 8; ++r) c[r] = b0;
#pragma unroll
    for (int k0 = 0; k0 < 192; k0 += 32)
      c = wmma16(afrag_f16(Mh, 192, k0), bfrag_pre(pWnode, 6, k0 >> 5, nt), c);
#pragma unroll
    for (int r = 0; r < 8; ++r) {
      int row = n0 + r + half * 8;
      if (row < N) h_ws[(long)row * 512 + o0 + nn] = c[r];
    }
  }
}

// hout[dst] += h[src] * a   (a = exp/ssum), float4 granularity
__global__ void k_scatter(const int* __restrict__ src, const int* __restrict__ dst,
                          const float* __restrict__ h_ws,
                          const float* __restrict__ scores,
                          const float* __restrict__ ssum, float* __restrict__ hout,
                          int E) {
  long idx = (long)blockIdx.x * blockDim.x + threadIdx.x;
  if (idx >= (long)E * 128) return;
  int e = (int)(idx >> 7), q = (int)(idx & 127), h = q >> 5;
  int d = dst[e];
  float a = scores[(long)e * 4 + h] / ssum[(long)d * 4 + h];
  float4 v = ((const float4*)h_ws)[(long)src[e] * 128 + q];
  float* o = &hout[(long)d * 512 + q * 4];
  atomicAdd(o + 0, v.x * a); atomicAdd(o + 1, v.y * a);
  atomicAdd(o + 2, v.z * a); atomicAdd(o + 3, v.w * a);
}

// final: leaky + head-sum, then on-KAN as K=1024 GEMM
__global__ __launch_bounds__(256) void k_out_node(
    const float* __restrict__ hout, const _Float16* __restrict__ pOn0,
    const _Float16* __restrict__ pOn1, const float* __restrict__ onb,
    float* __restrict__ out_node, int N) {
  __shared__ float HO[16 * 128];
  __shared__ _Float16 AK[16 * 1024];
  int n0 = blockIdx.x * 16, tid = threadIdx.x;
  for (int idx = tid; idx < 16 * 128; idx += 256) {
    int r = idx >> 7, c = idx & 127;
    int n = n0 + r; if (n >= N) n = N - 1;
    float s = 0.f;
#pragma unroll
    for (int h = 0; h < 4; ++h) s += leaky(hout[(long)n * 512 + h * 128 + c]);
    HO[idx] = s;
  }
  __syncthreads();
  for (int idx = tid; idx < 16 * 1024; idx += 256) {
    int r = idx >> 10, k = idx & 1023;
    int kk = k & 511, i = kk >> 2, g = kk & 3;
    float s, cv; __sincosf(HO[r * 128 + i] * (float)(g + 1), &s, &cv);
    AK[idx] = (_Float16)((k < 512) ? cv : s);
  }
  __syncthreads();
  int w = tid >> 5, o0 = w * 16, lane = tid & 31, nn = lane & 15, half = lane >> 4;
  float b0 = onb[o0 + nn];
  v8f c;
#pragma unroll
  for (int r = 0; r < 8; ++r) c[r] = b0;
#pragma unroll
  for (int k0 = 0; k0 < 1024; k0 += 32) {
    const _Float16* B = (k0 < 512) ? pOn0 : pOn1;
    c = wmma16(afrag_f16(AK, 1024, k0), bfrag_pre(B, 16, (k0 & 511) >> 5, w), c);
  }
#pragma unroll
  for (int r = 0; r < 8; ++r) {
    int row = n0 + r + half * 8;
    if (row < N) out_node[(long)row * 128 + o0 + nn] = c[r];
  }
}

// ---------------------------------------------------------------- launcher
extern "C" void kernel_launch(void* const* d_in, const int* in_sizes, int n_in,
                              void* d_out, int out_size, void* d_ws,
                              size_t ws_size, hipStream_t stream) {
  const float* node_feats = (const float*)d_in[0];
  const float* edge_feats = (const float*)d_in[1];
  const int* src = (const int*)d_in[2];
  const int* dst = (const int*)d_in[3];
  const float* W_ni = (const float*)d_in[4];
  const float* W_nj = (const float*)d_in[5];
  const float* W_fij = (const float*)d_in[6];
  const float* W_node = (const float*)d_in[7];
  const float* b_node = (const float*)d_in[8];
  const float* attn = (const float*)d_in[9];
  const float* bias_edge = (const float*)d_in[10];
  const float* ek_c = (const float*)d_in[11];
  const float* ek_b = (const float*)d_in[12];
  const float* nk_c = (const float*)d_in[13];
  const float* nk_b = (const float*)d_in[14];
  const float* on_c = (const float*)d_in[15];
  const float* on_b = (const float*)d_in[16];
  const float* oe_c = (const float*)d_in[17];
  const float* oe_b = (const float*)d_in[18];
  (void)n_in; (void)out_size; (void)ws_size;

  int N = in_sizes[0] / 128;
  int E = in_sizes[2];
  float* out_node = (float*)d_out;
  float* out_edge = out_node + (long)N * 128;

  float* ws = (float*)d_ws;
  float* f_ni = ws;                ws += (long)N * 128;
  float* f_nj = ws;                ws += (long)N * 128;
  float* scores = ws;              ws += (long)E * 4;
  float* zbase = ws;               // zeroed region starts here
  unsigned* smaxU = (unsigned*)ws; ws += (long)N * 4;
  float* ssum = ws;                ws += (long)N * 4;
  float* agg = ws;                 ws += (long)N * 64;
  float* deg = ws;                 ws += (long)N;
  float* hout = ws;                ws += (long)N * 512;
  float* h_ws = ws;                ws += (long)N * 512;

  // f16 pre-swizzled weight fragments (32B aligned)
  uintptr_t pa = ((uintptr_t)ws + 31) & ~(uintptr_t)31;
  _Float16* hw = (_Float16*)pa;
  _Float16* pWni = hw;   hw += 128 * 128;
  _Float16* pWnj = hw;   hw += 128 * 128;
  _Float16* pWfij = hw;  hw += 64 * 128;
  _Float16* pWnode = hw; hw += 192 * 512;
  _Float16* pEk0 = hw;   hw += 128 * 128;
  _Float16* pEk1 = hw;   hw += 128 * 128;
  _Float16* pNk0 = hw;   hw += 192 * 192;
  _Float16* pNk1 = hw;   hw += 192 * 192;
  _Float16* pOe0 = hw;   hw += 128 * 32;
  _Float16* pOe1 = hw;   hw += 128 * 32;
  _Float16* pOn0 = hw;   hw += 512 * 128;
  _Float16* pOn1 = hw;   hw += 512 * 128;

  auto prep = [&](const float* s, _Float16* d, int K, int Nn, int ld, int cm) {
    long total = (long)(K / 32) * (Nn / 16) * 512;
    k_prep<<<(unsigned)((total + 255) / 256), 256, 0, stream>>>(s, d, K / 32,
                                                                Nn / 16, ld, cm);
  };
  prep(W_ni, pWni, 128, 128, 128, 0);
  prep(W_nj, pWnj, 128, 128, 128, 0);
  prep(W_fij, pWfij, 64, 128, 128, 0);
  prep(W_node, pWnode, 192, 512, 512, 0);
  prep(ek_c, pEk0, 128, 128, 128, 1);
  prep(ek_c + 128 * 128, pEk1, 128, 128, 128, 1);
  prep(nk_c, pNk0, 192, 192, 192, 1);
  prep(nk_c + 192 * 192, pNk1, 192, 192, 192, 1);
  prep(oe_c, pOe0, 128, 32, 128, 1);
  prep(oe_c + 32 * 128, pOe1, 128, 32, 128, 1);
  prep(on_c, pOn0, 512, 128, 512, 1);
  prep(on_c + 128 * 512, pOn1, 512, 128, 512, 1);

  long nz = (long)N * (4 + 4 + 64 + 1 + 512);
  k_zero<<<(unsigned)((nz + 255) / 256), 256, 0, stream>>>(zbase, nz);

  k_node_lin<<<(N + 15) / 16, 256, 0, stream>>>(node_feats, pWni, pWnj, f_ni,
                                                f_nj, N);
  {
    long t = (long)E * 16;
    k_agg<<<(unsigned)((t + 255) / 256), 256, 0, stream>>>(edge_feats, dst, agg,
                                                           deg, E);
  }
  k_edge<<<(E + 15) / 16, 256, 0, stream>>>(
      edge_feats, src, dst, f_ni, f_nj, pWfij, pEk0, pEk1, ek_b, bias_edge,
      attn, pOe0, pOe1, oe_b, scores, smaxU, out_edge, E);
  {
    long t = (long)E * 4;
    k_softmax2<<<(unsigned)((t + 255) / 256), 256, 0, stream>>>(dst, smaxU,
                                                                scores, ssum, E);
  }
  k_node_kan<<<(N + 15) / 16, 256, 0, stream>>>(node_feats, agg, deg, pNk0,
                                                pNk1, nk_b, pWnode, b_node,
                                                h_ws, N);
  {
    long t = (long)E * 128;
    k_scatter<<<(unsigned)((t + 255) / 256), 256, 0, stream>>>(
        src, dst, h_ws, scores, ssum, hout, E);
  }
  k_out_node<<<(N + 15) / 16, 256, 0, stream>>>(hout, pOn0, pOn1, on_b,
                                                out_node, N);
}
